// ALiBiMultiHeadAttention_43791486550281
// MI455X (gfx1250) — compile-verified, matrix operand order checked by reading, not run on hardware
//
#include <hip/hip_runtime.h>
#include <math.h>
#include <stdint.h>

typedef __bf16 bf16_t;
typedef __attribute__((ext_vector_type(16))) __bf16 v16bf;
typedef __attribute__((ext_vector_type(8)))  __bf16 v8bf;
typedef __attribute__((ext_vector_type(8)))  float  v8f;
typedef __attribute__((ext_vector_type(4)))  unsigned int v4u;
typedef __attribute__((ext_vector_type(8)))  int v8i;
typedef __attribute__((ext_vector_type(4)))  int v4i;

#define EMBED_DIM 1024
#define NUM_HEADS 16
#define HEAD_DIM  64
#define BATCH     2
#define SEQ       2048
#define MTOK      (BATCH*SEQ)   /* 4096 */

__device__ __forceinline__ v8f wmma_bf16(v16bf a, v16bf b, v8f c) {
  // D(16x16,f32) = A(16x32,bf16) x B(32x16,bf16) + C
  return __builtin_amdgcn_wmma_f32_16x16x32_bf16(
      false, a, false, b, (short)0, c, false, false);
}

// Load one 16x32 bf16 fragment in the ISA A-layout (also used for B where the
// source matrix's rows run over the N dimension, contiguous along K).
// Per ISA 7.12.2: lanes 0-15 row r hold K=[0..7],[16..23]; lanes 16-31 hold
// K=[8..15],[24..31]. Two 16-byte loads per lane.
__device__ __forceinline__ v16bf load_frag(const bf16_t* __restrict__ src, int ld) {
  const int lane = threadIdx.x & 31;
  const int r    = lane & 15;
  const int kb   = (lane >> 4) << 3;      // 0 or 8
  const bf16_t* p = src + (size_t)r * ld + kb;
  v8bf lo = *(const v8bf*)(p);            // K = kb .. kb+7
  v8bf hi = *(const v8bf*)(p + 16);       // K = 16+kb .. 23+kb
  v16bf f;
#pragma unroll
  for (int i = 0; i < 8; ++i) { f[i] = lo[i]; f[8 + i] = hi[i]; }
  return f;
}

// Issue one TDM descriptor: copy a 64(N-rows) x 32(K) bf16 tile of a row-major
// [Nrows x 1024] matrix into LDS (row-major 64x32, ready for load_frag ld=32).
// D# per CDNA5 ISA: group0 = {count|flags, lds_addr,
// global_addr lo, global_addr hi | type=2}; group1 packs data_size=2B,
// tensor_dim0=1024, tensor_dim1=Nrows, tile_dim0=32, tile_dim1=64,
// tensor_dim0_stride=1024.  Groups 2/3 unused (2D tensor).
__device__ __forceinline__ void tdm_load_tile(const bf16_t* gsrc, unsigned lds_byte_off,
                                              int nrows) {
  uint64_t ga = (uint64_t)(uintptr_t)gsrc;
  v4u g0 = { 1u,                                   // count=1, user descriptor
             lds_byte_off,                         // lds_addr
             (unsigned)ga,                         // global_addr[31:0]
             (unsigned)((ga >> 32) & 0x01FFFFFFu) | 0x80000000u }; // [56:32]|type=2
  v8i g1 = { (int)0x00010000,                      // data_size=1 (2 bytes)
             (int)0x04000000,                      // tensor_dim0=1024 in bits[63:48]
             (int)((nrows & 0xFFFF) << 16),        // tensor_dim1 lo16 in bits[95:80]
             (int)(32 << 16),                      // tile_dim0=32 in bits[127:112]
             (int)64,                              // tile_dim1=64 in bits[143:128]
             (int)1024,                            // tensor_dim0_stride in bits[191:160]
             0, 0 };
  v4i gz4 = { 0, 0, 0, 0 };
  v8i gz8 = { 0, 0, 0, 0, 0, 0, 0, 0 };
  // 6-arg form (clang-23 / therock-10.0 headers): (g0, g1, g2, g3, g4, cpol)
  __builtin_amdgcn_tensor_load_to_lds(g0, g1, gz4, gz4, gz8, 0);
}

// ---------------------------------------------------------------------------
// fp32 -> bf16 conversion
// ---------------------------------------------------------------------------
__global__ void cvt_f32_to_bf16(const float* __restrict__ in,
                                bf16_t* __restrict__ out, int n) {
  int i = blockIdx.x * blockDim.x + threadIdx.x;
  int stride = gridDim.x * blockDim.x;
  for (; i < n; i += stride) out[i] = (bf16_t)in[i];
}

// ---------------------------------------------------------------------------
// QKV projection: [4096,1024] x [1024,3072] -> Q,K ([B,H,S,Hd]) and V^T
// ([B,H,Hd,S]).  Block = 8 waves sharing one TDM-staged 64x32 weight tile
// (double-buffered in LDS); each wave owns a 16-token A stripe.
// ---------------------------------------------------------------------------
__global__ void __launch_bounds__(256)
qkv_gemm(const bf16_t* __restrict__ xb, const bf16_t* __restrict__ wqkv,
         const float* __restrict__ b_qkv,
         bf16_t* __restrict__ qws, bf16_t* __restrict__ kws,
         bf16_t* __restrict__ vtws) {
  __shared__ bf16_t bsm[2][64 * 32];
  const int lane = threadIdx.x & 31;
  const int wave = threadIdx.x >> 5;
  const int mg = blockIdx.x & 31;        // 32 groups of 8 M-tiles (128 tokens)
  const int n4 = blockIdx.x >> 5;        // 48 groups of 64 output dims
  const int mt = mg * 8 + wave;
  const bool issuer = (wave == 0);
  const unsigned lds0 = (unsigned)(uintptr_t)&bsm[0][0];
  const bf16_t* wtile = wqkv + (size_t)(n4 * 64) * EMBED_DIM;

  v8f acc[4] = {};
  const bf16_t* arow = xb + (size_t)(mt * 16) * EMBED_DIM;

  if (issuer) tdm_load_tile(wtile, lds0, 3 * EMBED_DIM);

  for (int it = 0; it < EMBED_DIM / 32; ++it) {
    const int cur = it & 1;
    v16bf a = load_frag(arow + it * 32, EMBED_DIM);   // overlaps wait+barrier
    if (issuer) __builtin_amdgcn_s_wait_tensorcnt(0);
    __syncthreads();
    if (issuer && it + 1 < EMBED_DIM / 32)
      tdm_load_tile(wtile + (it + 1) * 32, lds0 + (unsigned)((cur ^ 1) * 64 * 32 * 2),
                    3 * EMBED_DIM);
#pragma unroll
    for (int nt = 0; nt < 4; ++nt) {
      v16bf b = load_frag(&bsm[cur][nt * 16 * 32], 32);
      acc[nt] = wmma_bf16(a, b, acc[nt]);
    }
    __syncthreads();
  }

  const int hb = (lane >> 4) << 3;
  const int nn = lane & 15;
#pragma unroll
  for (int nt = 0; nt < 4; ++nt) {
#pragma unroll
    for (int e = 0; e < 8; ++e) {
      int row = mt * 16 + hb + e;            // token index in [0, B*S)
      int o   = n4 * 64 + nt * 16 + nn;      // output dim in [0, 3D)
      float v = acc[nt][e] + b_qkv[o];
      bf16_t bv = (bf16_t)v;
      int three = o >> 10;
      int h  = (o >> 6) & (NUM_HEADS - 1);
      int hd = o & (HEAD_DIM - 1);
      int bb = row >> 11;
      int s  = row & (SEQ - 1);
      size_t bh = (size_t)bb * NUM_HEADS + h;
      if (three == 0)      qws[(bh * SEQ + s) * HEAD_DIM + hd] = bv;
      else if (three == 1) kws[(bh * SEQ + s) * HEAD_DIM + hd] = bv;
      else                 vtws[(bh * HEAD_DIM + hd) * SEQ + s] = bv;
    }
  }
}

// ---------------------------------------------------------------------------
// ALiBi flash attention.  One wave per (b, h, 16-query tile); 32 keys/iter:
// 4 WMMAs for scores, online softmax with half-wave shuffle reductions,
// P transposed through LDS, 4 WMMAs for P*V.
// ---------------------------------------------------------------------------
__global__ void __launch_bounds__(128)
alibi_attn(const bf16_t* __restrict__ qws, const bf16_t* __restrict__ kws,
           const bf16_t* __restrict__ vtws, bf16_t* __restrict__ attn) {
  __shared__ bf16_t smem[4 * 16 * 32];     // one 16x32 P tile per wave

  const int lane = threadIdx.x & 31;
  const int w    = threadIdx.x >> 5;
  const int gid  = blockIdx.x * 4 + w;
  const int qt = gid & 127;                // S/16 = 128 query tiles
  const int h  = (gid >> 7) & (NUM_HEADS - 1);
  const int b  = gid >> 11;

  const float slope = exp2f(-0.5f * (float)h);   // 2^(-8h/H), H=16
  const float scale = 0.125f;                    // 1/sqrt(64)

  const size_t bh = (size_t)b * NUM_HEADS + h;
  const bf16_t* qbase = qws + (bh * SEQ + (size_t)qt * 16) * HEAD_DIM;
  const bf16_t* kbase = kws + bh * SEQ * HEAD_DIM;
  const bf16_t* vbase = vtws + bh * (size_t)HEAD_DIM * SEQ;
  bf16_t* pst = smem + w * (16 * 32);

  v16bf qa0 = load_frag(qbase, HEAD_DIM);
  v16bf qa1 = load_frag(qbase + 32, HEAD_DIM);

  const int hb = (lane >> 4) << 3;
  const int nn = lane & 15;

  float mrow[8], lrow[8];
  v8f o0 = {}, o1 = {}, o2 = {}, o3 = {};
#pragma unroll
  for (int e = 0; e < 8; ++e) { mrow[e] = -INFINITY; lrow[e] = 0.f; }

  for (int kk = 0; kk < SEQ; kk += 32) {
    v8f s0 = {}, s1 = {};
    {
      const bf16_t* kp0 = kbase + (size_t)kk * HEAD_DIM;
      const bf16_t* kp1 = kbase + (size_t)(kk + 16) * HEAD_DIM;
      s0 = wmma_bf16(qa0, load_frag(kp0, HEAD_DIM), s0);
      s0 = wmma_bf16(qa1, load_frag(kp0 + 32, HEAD_DIM), s0);
      s1 = wmma_bf16(qa0, load_frag(kp1, HEAD_DIM), s1);
      s1 = wmma_bf16(qa1, load_frag(kp1 + 32, HEAD_DIM), s1);
    }

    float fac[8];
#pragma unroll
    for (int e = 0; e < 8; ++e) {
      int q = qt * 16 + hb + e;
      float a0 = s0[e] * scale - slope * fabsf((float)(q - (kk + nn)));
      float a1 = s1[e] * scale - slope * fabsf((float)(q - (kk + 16 + nn)));
      float rm = fmaxf(a0, a1);
#pragma unroll
      for (int msk = 1; msk < 16; msk <<= 1)
        rm = fmaxf(rm, __shfl_xor(rm, msk, 32));
      float nm = fmaxf(mrow[e], rm);
      fac[e] = __expf(mrow[e] - nm);
      float p0 = __expf(a0 - nm);
      float p1 = __expf(a1 - nm);
      float ps = p0 + p1;
#pragma unroll
      for (int msk = 1; msk < 16; msk <<= 1)
        ps += __shfl_xor(ps, msk, 32);
      lrow[e] = lrow[e] * fac[e] + ps;
      mrow[e] = nm;
      pst[(hb + e) * 32 + nn]      = (bf16_t)p0;
      pst[(hb + e) * 32 + 16 + nn] = (bf16_t)p1;
    }
#pragma unroll
    for (int e = 0; e < 8; ++e) {
      o0[e] *= fac[e]; o1[e] *= fac[e]; o2[e] *= fac[e]; o3[e] *= fac[e];
    }
    v16bf pa = load_frag(pst, 32);
    o0 = wmma_bf16(pa, load_frag(vbase + (size_t)0 * 16 * SEQ + kk, SEQ), o0);
    o1 = wmma_bf16(pa, load_frag(vbase + (size_t)1 * 16 * SEQ + kk, SEQ), o1);
    o2 = wmma_bf16(pa, load_frag(vbase + (size_t)2 * 16 * SEQ + kk, SEQ), o2);
    o3 = wmma_bf16(pa, load_frag(vbase + (size_t)3 * 16 * SEQ + kk, SEQ), o3);
  }

#pragma unroll
  for (int e = 0; e < 8; ++e) {
    int tok = qt * 16 + hb + e;
    float inv = 1.0f / lrow[e];
    size_t base = ((size_t)b * SEQ + tok) * EMBED_DIM + h * HEAD_DIM + nn;
    attn[base + 0]  = (bf16_t)(o0[e] * inv);
    attn[base + 16] = (bf16_t)(o1[e] * inv);
    attn[base + 32] = (bf16_t)(o2[e] * inv);
    attn[base + 48] = (bf16_t)(o3[e] * inv);
  }
}

// ---------------------------------------------------------------------------
// Output projection: [4096,1024] x [1024,1024] + bias -> fp32 out.
// Same TDM-staged cooperative structure as qkv_gemm.
// ---------------------------------------------------------------------------
__global__ void __launch_bounds__(256)
out_gemm(const bf16_t* __restrict__ attn, const bf16_t* __restrict__ wout,
         const float* __restrict__ b_out, float* __restrict__ out) {
  __shared__ bf16_t bsm[2][64 * 32];
  const int lane = threadIdx.x & 31;
  const int wave = threadIdx.x >> 5;
  const int mg = blockIdx.x & 31;        // 32 groups of 8 M-tiles
  const int n4 = blockIdx.x >> 5;        // 16 groups of 64 output dims
  const int mt = mg * 8 + wave;
  const bool issuer = (wave == 0);
  const unsigned lds0 = (unsigned)(uintptr_t)&bsm[0][0];
  const bf16_t* wtile = wout + (size_t)(n4 * 64) * EMBED_DIM;

  v8f acc[4] = {};
  const bf16_t* arow = attn + (size_t)(mt * 16) * EMBED_DIM;

  if (issuer) tdm_load_tile(wtile, lds0, EMBED_DIM);

  for (int it = 0; it < EMBED_DIM / 32; ++it) {
    const int cur = it & 1;
    v16bf a = load_frag(arow + it * 32, EMBED_DIM);
    if (issuer) __builtin_amdgcn_s_wait_tensorcnt(0);
    __syncthreads();
    if (issuer && it + 1 < EMBED_DIM / 32)
      tdm_load_tile(wtile + (it + 1) * 32, lds0 + (unsigned)((cur ^ 1) * 64 * 32 * 2),
                    EMBED_DIM);
#pragma unroll
    for (int nt = 0; nt < 4; ++nt) {
      v16bf b = load_frag(&bsm[cur][nt * 16 * 32], 32);
      acc[nt] = wmma_bf16(a, b, acc[nt]);
    }
    __syncthreads();
  }

  const int hb = (lane >> 4) << 3;
  const int nn = lane & 15;
#pragma unroll
  for (int nt = 0; nt < 4; ++nt) {
#pragma unroll
    for (int e = 0; e < 8; ++e) {
      int row = mt * 16 + hb + e;
      int o   = n4 * 64 + nt * 16 + nn;
      out[(size_t)row * EMBED_DIM + o] = acc[nt][e] + b_out[o];
    }
  }
}

// ---------------------------------------------------------------------------
extern "C" void kernel_launch(void* const* d_in, const int* in_sizes, int n_in,
                              void* d_out, int out_size, void* d_ws, size_t ws_size,
                              hipStream_t stream) {
  const float* x     = (const float*)d_in[0];
  const float* W_qkv = (const float*)d_in[1];
  const float* b_qkv = (const float*)d_in[2];
  const float* W_out = (const float*)d_in[3];
  const float* b_out = (const float*)d_in[4];
  float* out = (float*)d_out;

  // Workspace carve-up (bf16 elements), ~48 MB total.
  bf16_t* xb    = (bf16_t*)d_ws;
  bf16_t* wqkvb = xb    + (size_t)MTOK * EMBED_DIM;
  bf16_t* woutb = wqkvb + (size_t)3 * EMBED_DIM * EMBED_DIM;
  bf16_t* qws   = woutb + (size_t)EMBED_DIM * EMBED_DIM;
  bf16_t* kws   = qws   + (size_t)MTOK * EMBED_DIM;
  bf16_t* vtws  = kws   + (size_t)MTOK * EMBED_DIM;
  bf16_t* attn  = vtws  + (size_t)MTOK * EMBED_DIM;

  const int nx = MTOK * EMBED_DIM;
  cvt_f32_to_bf16<<<1024, 256, 0, stream>>>(x, xb, nx);
  cvt_f32_to_bf16<<<1024, 256, 0, stream>>>(W_qkv, wqkvb, 3 * EMBED_DIM * EMBED_DIM);
  cvt_f32_to_bf16<<<1024, 256, 0, stream>>>(W_out, woutb, EMBED_DIM * EMBED_DIM);

  // 32 M-groups x 48 N-quads, 8 waves/block
  qkv_gemm<<<1536, 256, 0, stream>>>(xb, wqkvb, b_qkv, qws, kws, vtws);
  // B*H*(S/16) = 4096 waves, 4 waves/block
  alibi_attn<<<1024, 128, 0, stream>>>(qws, kws, vtws, attn);
  // 32 M-groups x 16 N-quads
  out_gemm<<<512, 256, 0, stream>>>(attn, woutb, b_out, out);
}
